// DualPathAttentionDownSamplingModule_4466765988510
// MI455X (gfx1250) — matrix operand order repaired and checked by reading, not compiled, not run on hardware
//
#include <hip/hip_runtime.h>

typedef __attribute__((ext_vector_type(16))) _Float16 v16h;
typedef __attribute__((ext_vector_type(8)))  float    v8f;

#define BB 4
#define CIN 256
#define NN 3136      // 56*56
#define N2 784       // 28*28
#define GN_EPS 1e-5f

union Frag {
  v16h v;
  _Float16 h[16];
  uint4 q4[2];
};

__device__ inline void frag_zero(Frag& f) {
  f.q4[0] = make_uint4(0u,0u,0u,0u);
  f.q4[1] = make_uint4(0u,0u,0u,0u);
}
// A-matrix 16x32 f16 layout: lane m=l&15, K=(j&7)+(j>>3)*16+hi*8
__device__ inline void load_a_frag(Frag& f, const _Float16* row, int hi) {
  f.q4[0] = *(const uint4*)(row + hi*8);
  f.q4[1] = *(const uint4*)(row + 16 + hi*8);
}
// B-matrix 32x16 f16 layout: lane n=l&15, K=(j&7)+(j>>3)*16+hi*8
__device__ inline void load_b_frag(Frag& f, const _Float16* row, int hi) {
  f.q4[0] = *(const uint4*)(row + hi*8);
  f.q4[1] = *(const uint4*)(row + 16 + hi*8);
}
__device__ inline v8f wmma_f16(const Frag& a, const Frag& b, v8f c) {
  return __builtin_amdgcn_wmma_f32_16x16x32_f16(false, a.v, false, b.v, (short)0, c, false, false);
}

// ---------------- elementwise / prep kernels ----------------

__global__ void k_cvt_f16(const float* __restrict__ s, _Float16* __restrict__ d, int n) {
  int i = blockIdx.x*256 + threadIdx.x;
  if (i < n) d[i] = (_Float16)s[i];
}

// proj_w (512,256,3,3) -> Wr[o][(ky*3+kx)*256 + c] f16
__global__ void k_repack_proj(const float* __restrict__ w, _Float16* __restrict__ d) {
  int i = blockIdx.x*256 + threadIdx.x;
  if (i >= 512*9*256) return;
  int c = i % 256; int p = (i/256)%9; int o = i/(256*9);
  d[(o*9+p)*256 + c] = (_Float16)w[(o*256 + c)*9 + p];
}

// mixed bias table: mb[i,p] = sum_h th1[i,h]*attn_biases[h,p]
__global__ void k_mbias(const float* __restrict__ th1, const float* __restrict__ ab,
                        float* __restrict__ mb) {
  int i = blockIdx.x*256 + threadIdx.x;
  if (i >= 8*NN) return;
  int p = i % NN; int hh = i / NN;
  float a = 0.f;
  #pragma unroll
  for (int h = 0; h < 8; ++h) a += th1[hh*8+h]*ab[h*NN+p];
  mb[i] = a;
}

// x (B,256,56,56) f32 -> xt[b][n][c] f16
__global__ void k_xt(const float* __restrict__ x, _Float16* __restrict__ xt) {
  int i = blockIdx.x*256 + threadIdx.x;
  if (i >= BB*CIN*NN) return;
  int n = i % NN; int c = (i/NN)%CIN; int b = i/(NN*CIN);
  xt[((long)b*NN+n)*CIN + c] = (_Float16)x[i];
}

// depthwise 3x3 stride2 pad1 + bias + pool(::2,::2) -> tt[b][q][c] f16
__global__ void k_qlocal(const float* __restrict__ x, const float* __restrict__ w,
                         const float* __restrict__ bias, _Float16* __restrict__ tt) {
  int i = blockIdx.x*256 + threadIdx.x;
  if (i >= BB*CIN*N2) return;
  int xx = i % 28; int yy = (i/28)%28; int c = (i/784)%CIN; int b = i/(784*CIN);
  const float* xp = x + ((long)b*CIN + c)*NN;
  float acc = bias[c];
  #pragma unroll
  for (int ky = 0; ky < 3; ++ky) {
    int sy = 2*yy - 1 + ky;
    if ((unsigned)sy >= 56u) continue;
    #pragma unroll
    for (int kx = 0; kx < 3; ++kx) {
      int sx = 2*xx - 1 + kx;
      if ((unsigned)sx >= 56u) continue;
      acc += xp[sy*56+sx]*w[c*9 + ky*3 + kx];
    }
  }
  acc += xp[(2*yy)*56 + 2*xx];   // pool_q
  tt[((long)b*N2 + yy*28 + xx)*CIN + c] = (_Float16)acc;
}

// ---------------- WMMA GEMM for 1x1 convs: Y[b,o,n] = W[o,:]·Xt[b,n,:] + bias ----------------

__global__ void k_gemm(const _Float16* __restrict__ W, const _Float16* __restrict__ Xt,
                       const float* __restrict__ bias, float* __restrict__ Y,
                       int Bn, int O, int C, int N) {
  int wid = threadIdx.x >> 5, lane = threadIdx.x & 31;
  int hi = lane >> 4, ln = lane & 15;
  int ntil = N/16;
  int tilesPerB = (O/16)*ntil;
  int gt = blockIdx.x*8 + wid;
  if (gt >= Bn*tilesPerB) return;
  int b = gt / tilesPerB;
  int r = gt % tilesPerB;
  int o0 = (r / ntil)*16;
  int n0 = (r % ntil)*16;
  const _Float16* wrow = W + (long)(o0 + ln)*C;
  const _Float16* xrow = Xt + ((long)b*N + n0 + ln)*C;
  v8f acc = {};
  for (int ck = 0; ck < C; ck += 32) {
    Frag a, bf;
    load_a_frag(a, wrow + ck, hi);
    load_b_frag(bf, xrow + ck, hi);
    acc = wmma_f16(a, bf, acc);
  }
  #pragma unroll
  for (int i = 0; i < 8; ++i) {
    int oo = o0 + i + hi*8;
    float v = acc[i];
    if (bias) v += bias[oo];
    Y[((long)b*O + oo)*N + n0 + ln] = v;
  }
}

// ---------------- WMMA implicit-GEMM 3x3 stride2 pad1 conv (proj path) ----------------

__global__ void k_conv3x3(const _Float16* __restrict__ Wr, const _Float16* __restrict__ xt,
                          const float* __restrict__ bias, float* __restrict__ Y) {
  int wid = threadIdx.x >> 5, lane = threadIdx.x & 31;
  int hi = lane >> 4, ln = lane & 15;
  const int tilesPerB = 32*49;
  int gt = blockIdx.x*8 + wid;
  if (gt >= BB*tilesPerB) return;
  int b = gt / tilesPerB; int r = gt % tilesPerB;
  int o0 = (r/49)*16, n0 = (r%49)*16;
  int n = n0 + ln;
  int oy = n/28, ox = n%28;
  const _Float16* wrow = Wr + (long)(o0+ln)*2304;
  v8f acc = {};
  for (int ck = 0; ck < 2304; ck += 32) {
    int plane = ck >> 8;           // (ky*3+kx)
    int c0 = ck & 255;
    int ky = plane/3, kx = plane%3;
    int sy = 2*oy - 1 + ky, sx = 2*ox - 1 + kx;
    Frag a, bf;
    load_a_frag(a, wrow + ck, hi);
    if ((unsigned)sy < 56u && (unsigned)sx < 56u) {
      load_b_frag(bf, xt + ((long)b*NN + sy*56+sx)*CIN + c0, hi);
    } else {
      frag_zero(bf);
    }
    // prefetch next k-step's activation row (next plane wraps cleanly)
    if (ck + 32 < 2304) {
      int pl2 = (ck+32) >> 8;
      int ky2 = pl2/3, kx2 = pl2%3;
      int sy2 = 2*oy - 1 + ky2, sx2 = 2*ox - 1 + kx2;
      if ((unsigned)sy2 < 56u && (unsigned)sx2 < 56u)
        __builtin_prefetch(xt + ((long)b*NN + sy2*56+sx2)*CIN + ((ck+32)&255), 0, 3);
    }
    acc = wmma_f16(a, bf, acc);
  }
  #pragma unroll
  for (int i = 0; i < 8; ++i) {
    int oo = o0 + i + hi*8;
    Y[((long)b*512 + oo)*784 + n] = acc[i] + bias[oo];
  }
}

// ---------------- GroupNorm (32 groups of 16 ch), in place on (B,512,N) ----------------

__global__ void k_gn(float* __restrict__ Y, const float* __restrict__ g,
                     const float* __restrict__ bt, int N) {
  __shared__ float s1[256], s2[256];
  int b = blockIdx.x >> 5;
  int grp = blockIdx.x & 31;
  float* base = Y + ((long)b*512 + grp*16)*N;
  int cnt = 16*N;
  float sum = 0.f, sq = 0.f;
  for (int i = threadIdx.x; i < cnt; i += 256) { float v = base[i]; sum += v; sq += v*v; }
  s1[threadIdx.x] = sum; s2[threadIdx.x] = sq;
  __syncthreads();
  for (int st = 128; st > 0; st >>= 1) {
    if (threadIdx.x < st) { s1[threadIdx.x] += s1[threadIdx.x+st]; s2[threadIdx.x] += s2[threadIdx.x+st]; }
    __syncthreads();
  }
  float mean = s1[0]/(float)cnt;
  float var  = s2[0]/(float)cnt - mean*mean;
  float inv  = rsqrtf(var + GN_EPS);
  for (int i = threadIdx.x; i < cnt; i += 256) {
    int c = grp*16 + i/N;
    base[i] = (base[i]-mean)*inv*g[c] + bt[c];
  }
}

// ---------------- l2norm across heads -> f16 transposed buffers ----------------

__global__ void k_l2q(const float* __restrict__ qY, _Float16* __restrict__ qn) {
  int i = blockIdx.x*256 + threadIdx.x;
  if (i >= BB*N2*64) return;
  int d = i % 64; int q = (i/64)%N2; int b = i/(64*N2);
  float s = 0.f; float vals[8];
  #pragma unroll
  for (int h = 0; h < 8; ++h) { float v = qY[((long)b*512 + h*64 + d)*N2 + q]; vals[h] = v; s += v*v; }
  float inv = 1.f / fmaxf(sqrtf(s), 1e-12f);
  #pragma unroll
  for (int h = 0; h < 8; ++h) qn[(((long)(b*8+h))*N2 + q)*64 + d] = (_Float16)(vals[h]*inv);
}

__global__ void k_l2k(const float* __restrict__ kY, _Float16* __restrict__ kn) {
  int i = blockIdx.x*256 + threadIdx.x;
  if (i >= BB*NN*64) return;
  int d = i % 64; int n = (i/64)%NN; int b = i/(64*NN);
  float s = 0.f; float vals[8];
  #pragma unroll
  for (int h = 0; h < 8; ++h) { float v = kY[((long)b*512 + h*64 + d)*NN + n]; vals[h] = v; s += v*v; }
  float inv = 1.f / fmaxf(sqrtf(s), 1e-12f);
  #pragma unroll
  for (int h = 0; h < 8; ++h) kn[(((long)(b*8+h))*NN + n)*64 + d] = (_Float16)(vals[h]*inv);
}

// ---------------- grouped 3x3 stride2 conv on v_map (groups=8) ----------------

__global__ void k_loc(const float* __restrict__ vY, const float* __restrict__ w,
                      const float* __restrict__ bias, float* __restrict__ out) {
  int i = blockIdx.x*256 + threadIdx.x;
  if (i >= BB*512*N2) return;
  int xx = i % 28; int yy = (i/28)%28; int oc = (i/784)%512; int b = i/(784*512);
  int g = oc >> 6;
  float acc = bias[oc];
  for (int ic = 0; ic < 64; ++ic) {
    const float* src = vY + ((long)b*512 + g*64 + ic)*NN;
    const float* wp = w + ((long)oc*64 + ic)*9;
    #pragma unroll
    for (int ky = 0; ky < 3; ++ky) {
      int sy = 2*yy - 1 + ky;
      if ((unsigned)sy >= 56u) continue;
      #pragma unroll
      for (int kx = 0; kx < 3; ++kx) {
        int sx = 2*xx - 1 + kx;
        if ((unsigned)sx >= 56u) continue;
        acc += src[sy*56+sx]*wp[ky*3+kx];
      }
    }
  }
  out[((long)b*512 + oc)*784 + yy*28 + xx] = acc;
}

// ---------------- attention phase A: per-row (max, sumexp) of th1-mixed logits ----------------
// block = (b, q-tile of 16); wave h computes S_h = qn_h @ kn_h^T (WMMA); cross-head mix via LDS.

__global__ void k_attn_stats(const _Float16* __restrict__ qn, const _Float16* __restrict__ kn,
                             const int* __restrict__ bidx, const float* __restrict__ mb,
                             const float* __restrict__ th1, float* __restrict__ stats) {
  __shared__ float S[8][16][32];
  __shared__ float th1s[64];
  int wid = threadIdx.x >> 5, lane = threadIdx.x & 31, hi = lane >> 4, ln = lane & 15;
  if (threadIdx.x < 64) th1s[threadIdx.x] = th1[threadIdx.x];
  int b = blockIdx.x / 49; int q0 = (blockIdx.x % 49)*16;
  const _Float16* arow  = qn + (((long)(b*8+wid))*N2 + q0 + ln)*64;
  const _Float16* kbase = kn + ((long)(b*8+wid))*NN*64;
  float m = -3.0e38f, s = 0.f;
  __syncthreads();
  for (int nc = 0; nc < 98; ++nc) {
    int n0 = nc*32;
    v8f c0 = {}, c1 = {};
    #pragma unroll
    for (int ks = 0; ks < 64; ks += 32) {
      Frag a, b0, b1;
      load_a_frag(a, arow + ks, hi);
      load_b_frag(b0, kbase + (long)(n0 + ln)*64 + ks, hi);
      c0 = wmma_f16(a, b0, c0);
      load_b_frag(b1, kbase + (long)(n0 + 16 + ln)*64 + ks, hi);
      c1 = wmma_f16(a, b1, c1);
    }
    // prefetch next chunk's kn rows behind the barriers (covers full 32x64 f16 tile)
    if (nc + 1 < 98) {
      __builtin_prefetch(kbase + (long)(n0 + 32 + ln)*64, 0, 3);
      __builtin_prefetch(kbase + (long)(n0 + 48 + ln)*64, 0, 3);
    }
    __syncthreads();                 // prior iteration done reading S
    #pragma unroll
    for (int i = 0; i < 8; ++i) {
      S[wid][i+hi*8][ln]    = c0[i];
      S[wid][i+hi*8][16+ln] = c1[i];
    }
    __syncthreads();
    const int* brow = bidx + (q0+ln)*NN + n0;
    const float* mbr = mb + wid*NN;
    for (int cc = 0; cc < 16; ++cc) {
      int col = hi*16 + cc;
      float a2 = 0.f;
      #pragma unroll
      for (int h = 0; h < 8; ++h) a2 += th1s[wid*8+h]*S[h][ln][col];
      float L = 0.125f*a2 + mbr[brow[col]];
      if (L > m) { s = s*expf(m - L) + 1.f; m = L; }
      else         s += expf(L - m);
    }
  }
  float mo = __shfl_xor(m, 16, 32);
  float so = __shfl_xor(s, 16, 32);
  float mm = fmaxf(m, mo);
  float ss = s*expf(m - mm) + so*expf(mo - mm);
  if (lane < 16) {
    long o = (((long)(b*8+wid))*N2 + q0 + ln)*2;
    stats[o] = mm; stats[o+1] = ss;
  }
}

// ---------------- attention phase B: P = softmax, th2-mix in LDS, O += P2 @ V (WMMA) ----------------

__global__ void k_attn_out(const _Float16* __restrict__ qn, const _Float16* __restrict__ kn,
                           const _Float16* __restrict__ v16, const int* __restrict__ bidx,
                           const float* __restrict__ mb, const float* __restrict__ th1,
                           const float* __restrict__ th2, const float* __restrict__ stats,
                           float* __restrict__ outp) {
  __shared__ float S[8][16][32];
  __shared__ float P[8][16][32];
  __shared__ float th1s[64], th2s[64];
  int wid = threadIdx.x >> 5, lane = threadIdx.x & 31, hi = lane >> 4, ln = lane & 15;
  if (threadIdx.x < 64) { th1s[threadIdx.x] = th1[threadIdx.x]; th2s[threadIdx.x] = th2[threadIdx.x]; }
  int b = blockIdx.x / 49; int q0 = (blockIdx.x % 49)*16;
  const _Float16* arow  = qn + (((long)(b*8+wid))*N2 + q0 + ln)*64;
  const _Float16* kbase = kn + ((long)(b*8+wid))*NN*64;
  const _Float16* vbase = v16 + ((long)b*512 + wid*64)*NN;
  long so_ = (((long)(b*8+wid))*N2 + q0 + ln)*2;
  float ms = stats[so_];
  float sinv = 1.f / stats[so_+1];
  v8f zero = {};
  v8f acc[4] = {zero, zero, zero, zero};
  __syncthreads();
  for (int nc = 0; nc < 98; ++nc) {
    int n0 = nc*32;
    v8f c0 = {}, c1 = {};
    #pragma unroll
    for (int ks = 0; ks < 64; ks += 32) {
      Frag a, b0, b1;
      load_a_frag(a, arow + ks, hi);
      load_b_frag(b0, kbase + (long)(n0 + ln)*64 + ks, hi);
      c0 = wmma_f16(a, b0, c0);
      load_b_frag(b1, kbase + (long)(n0 + 16 + ln)*64 + ks, hi);
      c1 = wmma_f16(a, b1, c1);
    }
    // prefetch next chunk's kn and v rows behind the barriers
    if (nc + 1 < 98) {
      __builtin_prefetch(kbase + (long)(n0 + 32 + ln)*64, 0, 3);
      __builtin_prefetch(kbase + (long)(n0 + 48 + ln)*64, 0, 3);
      __builtin_prefetch(vbase + (long)(ln*4 + hi)*NN + n0 + 32, 0, 3);  // 32 rows of v, next chunk
    }
    __syncthreads();                 // prior iteration done reading S and P
    #pragma unroll
    for (int i = 0; i < 8; ++i) {
      S[wid][i+hi*8][ln]    = c0[i];
      S[wid][i+hi*8][16+ln] = c1[i];
    }
    __syncthreads();
    const int* brow = bidx + (q0+ln)*NN + n0;
    const float* mbr = mb + wid*NN;
    for (int cc = 0; cc < 16; ++cc) {
      int col = hi*16 + cc;
      float a2 = 0.f;
      #pragma unroll
      for (int h = 0; h < 8; ++h) a2 += th1s[wid*8+h]*S[h][ln][col];
      float L = 0.125f*a2 + mbr[brow[col]];
      P[wid][ln][col] = expf(L - ms)*sinv;
    }
    __syncthreads();
    Frag pa;                          // A-fragment of th2-mixed probs (16q x 32n)
    #pragma unroll
    for (int j = 0; j < 16; ++j) {
      int k = (j & 7) + ((j >> 3) << 4) + hi*8;
      float a2 = 0.f;
      #pragma unroll
      for (int i2 = 0; i2 < 8; ++i2) a2 += th2s[wid*8+i2]*P[i2][ln][k];
      pa.h[j] = (_Float16)a2;
    }
    #pragma unroll
    for (int dt = 0; dt < 4; ++dt) {
      Frag vb;
      load_b_frag(vb, vbase + (long)(dt*16 + ln)*NN + n0, hi);
      acc[dt] = wmma_f16(pa, vb, acc[dt]);
    }
  }
  #pragma unroll
  for (int dt = 0; dt < 4; ++dt)
    #pragma unroll
    for (int i = 0; i < 8; ++i)
      outp[((long)b*512 + wid*64 + dt*16 + ln)*784 + q0 + i + hi*8] = acc[dt][i];
}

// ---------------- hardswish(out_attn + v_local) -> transposed f16 for out-conv GEMM ----------------

__global__ void k_hsw(const float* __restrict__ op, const float* __restrict__ vl,
                      _Float16* __restrict__ ht) {
  int i = blockIdx.x*256 + threadIdx.x;
  if (i >= BB*512*N2) return;
  int q = i % 784; int c = (i/784)%512; int b = i/(784*512);
  float v = op[i] + vl[i];
  float h = v * fminf(fmaxf(v + 3.f, 0.f), 6.f) * (1.f/6.f);
  ht[((long)b*784 + q)*512 + c] = (_Float16)h;
}

__global__ void k_final(const float* __restrict__ oc, const float* __restrict__ xp,
                        float* __restrict__ out) {
  int i = blockIdx.x*256 + threadIdx.x;
  if (i < BB*512*N2) out[i] = oc[i] + xp[i];
}

// ---------------- host launch ----------------

extern "C" void kernel_launch(void* const* d_in, const int* in_sizes, int n_in,
                              void* d_out, int out_size, void* d_ws, size_t ws_size,
                              hipStream_t stream) {
  const float* x     = (const float*)d_in[0];
  const float* qlw   = (const float*)d_in[1];
  const float* qlb   = (const float*)d_in[2];
  const float* qpw   = (const float*)d_in[3];
  const float* qpb   = (const float*)d_in[4];
  const float* qgnw  = (const float*)d_in[5];
  const float* qgnb  = (const float*)d_in[6];
  const float* kw    = (const float*)d_in[7];
  const float* kgnw  = (const float*)d_in[8];
  const float* kgnb  = (const float*)d_in[9];
  const float* vw    = (const float*)d_in[10];
  const float* vgnw  = (const float*)d_in[11];
  const float* vgnb  = (const float*)d_in[12];
  const float* locw  = (const float*)d_in[13];
  const float* locb  = (const float*)d_in[14];
  const float* locgnw= (const float*)d_in[15];
  const float* locgnb= (const float*)d_in[16];
  const float* th1   = (const float*)d_in[17];
  const float* th2   = (const float*)d_in[18];
  const float* outw  = (const float*)d_in[19];
  const float* outgnw= (const float*)d_in[20];
  const float* outgnb= (const float*)d_in[21];
  const float* projw = (const float*)d_in[22];
  const float* projb = (const float*)d_in[23];
  const float* pgnw  = (const float*)d_in[24];
  const float* pgnb  = (const float*)d_in[25];
  const float* ab    = (const float*)d_in[26];
  const int*   bidx  = (const int*)d_in[27];

  char* wsb = (char*)d_ws;
  size_t off = 0;
  auto alloc = [&](size_t bytes) -> void* {
    void* p = wsb + off;
    off = (off + bytes + 255) & ~(size_t)255;
    return p;
  };
  _Float16* xt    = (_Float16*)alloc((size_t)BB*NN*CIN*2);
  _Float16* tt    = (_Float16*)alloc((size_t)BB*N2*CIN*2);
  _Float16* wq16  = (_Float16*)alloc((size_t)512*256*2);
  _Float16* wk16  = (_Float16*)alloc((size_t)512*256*2);
  _Float16* wv16  = (_Float16*)alloc((size_t)512*256*2);
  _Float16* wo16  = (_Float16*)alloc((size_t)512*512*2);
  _Float16* wp16  = (_Float16*)alloc((size_t)512*2304*2);
  float* mb       = (float*)alloc((size_t)8*NN*4);
  float* qY       = (float*)alloc((size_t)BB*512*N2*4);
  float* kY       = (float*)alloc((size_t)BB*512*NN*4);
  float* vY       = (float*)alloc((size_t)BB*512*NN*4);
  float* xprojY   = (float*)alloc((size_t)BB*512*N2*4);
  float* vlocY    = (float*)alloc((size_t)BB*512*N2*4);
  _Float16* qn16  = (_Float16*)alloc((size_t)BB*8*N2*64*2);
  _Float16* kn16  = (_Float16*)alloc((size_t)BB*8*NN*64*2);
  _Float16* v16b  = (_Float16*)alloc((size_t)BB*512*NN*2);
  float* stats    = (float*)alloc((size_t)BB*8*N2*2*4);
  float* outpre   = (float*)alloc((size_t)BB*512*N2*4);
  _Float16* ht    = (_Float16*)alloc((size_t)BB*N2*512*2);
  float* outc     = (float*)alloc((size_t)BB*512*N2*4);

  auto G = [](long n) { return (int)((n + 255)/256); };

  // weight conversion / repack, bias mix, input transposes
  k_cvt_f16<<<G(512*256), 256, 0, stream>>>(qpw, wq16, 512*256);
  k_cvt_f16<<<G(512*256), 256, 0, stream>>>(kw,  wk16, 512*256);
  k_cvt_f16<<<G(512*256), 256, 0, stream>>>(vw,  wv16, 512*256);
  k_cvt_f16<<<G(512*512), 256, 0, stream>>>(outw, wo16, 512*512);
  k_repack_proj<<<G(512*9*256), 256, 0, stream>>>(projw, wp16);
  k_mbias<<<G(8*NN), 256, 0, stream>>>(th1, ab, mb);
  k_xt<<<G((long)BB*CIN*NN), 256, 0, stream>>>(x, xt);
  k_qlocal<<<G((long)BB*CIN*N2), 256, 0, stream>>>(x, qlw, qlb, tt);

  // WMMA GEMMs (1x1 convs) + implicit-GEMM 3x3 proj conv
  k_gemm<<<BB*32*49/8,  256, 0, stream>>>(wq16, tt, qpb, qY, BB, 512, 256, N2);
  k_gemm<<<BB*32*196/8, 256, 0, stream>>>(wk16, xt, nullptr, kY, BB, 512, 256, NN);
  k_gemm<<<BB*32*196/8, 256, 0, stream>>>(wv16, xt, nullptr, vY, BB, 512, 256, NN);
  k_conv3x3<<<BB*32*49/8, 256, 0, stream>>>(wp16, xt, projb, xprojY);

  // group norms
  k_gn<<<BB*32, 256, 0, stream>>>(qY, qgnw, qgnb, N2);
  k_gn<<<BB*32, 256, 0, stream>>>(kY, kgnw, kgnb, NN);
  k_gn<<<BB*32, 256, 0, stream>>>(vY, vgnw, vgnb, NN);
  k_gn<<<BB*32, 256, 0, stream>>>(xprojY, pgnw, pgnb, N2);

  // l2norm across heads -> f16, v -> f16, local value conv + GN
  k_l2q<<<G((long)BB*N2*64), 256, 0, stream>>>(qY, qn16);
  k_l2k<<<G((long)BB*NN*64), 256, 0, stream>>>(kY, kn16);
  k_cvt_f16<<<G((long)BB*512*NN), 256, 0, stream>>>(vY, v16b, BB*512*NN);
  k_loc<<<G((long)BB*512*N2), 256, 0, stream>>>(vY, locw, locb, vlocY);
  k_gn<<<BB*32, 256, 0, stream>>>(vlocY, locgnw, locgnb, N2);

  // two-pass talking-heads attention (never materializes logits)
  k_attn_stats<<<BB*49, 256, 0, stream>>>(qn16, kn16, bidx, mb, th1, stats);
  k_attn_out<<<BB*49, 256, 0, stream>>>(qn16, kn16, v16b, bidx, mb, th1, th2, stats, outpre);

  // epilogue: hardswish -> out 1x1 conv (WMMA) -> GN -> + x_proj
  k_hsw<<<G((long)BB*512*N2), 256, 0, stream>>>(outpre, vlocY, ht);
  k_gemm<<<BB*32*49/8, 256, 0, stream>>>(wo16, ht, nullptr, outc, BB, 512, 512, N2);
  k_gn<<<BB*32, 256, 0, stream>>>(outc, outgnw, outgnb, N2);
  k_final<<<G((long)BB*512*N2), 256, 0, stream>>>(outc, xprojY, (float*)d_out);
}
